// EGNNLayerTorch_31653908971779
// MI455X (gfx1250) — matrix operand order repaired and checked by reading, not compiled, not run on hardware
//
#include <hip/hip_runtime.h>
#include <hip/hip_bf16.h>

typedef __attribute__((ext_vector_type(16))) _Float16 v16h;
typedef __attribute__((ext_vector_type(8)))  float    v8f;

#define DH 128
#define DM 128
#define DE 16

// fast silu: single v_exp_f32 + v_rcp_f32 + v_mul (no IEEE div expansion)
__device__ __forceinline__ float silu_f(float v) {
    return v * __builtin_amdgcn_rcpf(1.0f + __expf(-v));
}

__device__ __forceinline__ v8f wmma_f16(v16h a, v16h b, v8f c) {
    // D = A(16x32 f16) x B(32x16 f16) + C(16x16 f32)
    return __builtin_amdgcn_wmma_f32_16x16x32_f16(
        /*neg_a=*/false, a, /*neg_b=*/false, b,
        /*c_mod=*/(short)0, c, /*reuse_a=*/false, /*reuse_b=*/false);
}

// A fragment from LDS row-major f16 panel [16][pitch].
// Lane L holds row (L&15); K subrange: {kh*16..kh*16+15}
__device__ __forceinline__ v16h load_a_lds(const _Float16* __restrict__ As,
                                           int row, int pitch, int kbase, int kh) {
    v16h a;
    const _Float16* p0 = As + row * pitch + kbase + kh * 16;
#pragma unroll
    for (int i = 0; i < 16; ++i) { a[i] = p0[i]; }
    return a;
}

// B fragment from global f16 weight stored transposed: WT[n][kp]  (n = output col)
// Lane L: n = ntile*16 + (L&15); b[i] = B[kbase + kh*16 + i][n]  -> contiguous in WT
__device__ __forceinline__ v16h load_b_wt(const _Float16* __restrict__ WT,
                                          int n, int kp, int kbase, int kh) {
    v16h b;
    const _Float16* p = WT + (size_t)n * kp + kbase + kh * 16;
#pragma unroll
    for (int i = 0; i < 16; ++i) b[i] = p[i];
    return b;
}

// ---------------- prep: transpose fp32 [Kvalid][128] -> f16 [128][KP], zero pad K
__global__ void transpose_w_f16(_Float16* __restrict__ dst, const float* __restrict__ src,
                                int Kvalid, int KP, int total) {
    int t = blockIdx.x * blockDim.x + threadIdx.x;
    if (t >= total) return;
    int n = t / KP, k = t - n * KP;
    dst[t] = (k < Kvalid) ? (_Float16)src[(size_t)k * 128 + n] : (_Float16)0.0f;
}

__global__ void zero_f32(float* __restrict__ p, long long n) {
    long long i = (long long)blockIdx.x * blockDim.x + threadIdx.x;
    long long stride = (long long)gridDim.x * blockDim.x;
    for (; i < n; i += stride) p[i] = 0.0f;
}

// ---------------- edge kernel: one wave = 16 edges --------------------------
__global__ __launch_bounds__(32)
void egnn_edge_kernel(const float* __restrict__ h, const float* __restrict__ x,
                      const float* __restrict__ ea, const int* __restrict__ ei,
                      const float* __restrict__ be1, const float* __restrict__ be2,
                      const float* __restrict__ bx1, const float* __restrict__ Wx2,
                      const float* __restrict__ bx2,
                      const _Float16* __restrict__ We1T, const _Float16* __restrict__ We2T,
                      const _Float16* __restrict__ Wx1T,
                      float* __restrict__ agg, float* __restrict__ cnt,
                      float* __restrict__ coord, int E) {
    __shared__ _Float16 As[16 * 304];   // msg_in panel 16 x 288 (padded); reused for T
    __shared__ _Float16 M1[16 * 136];   // hidden1 16 x 128 f16
    __shared__ _Float16 MS[16 * 136];   // msg 16 x 128 f16
    __shared__ float    sdiff[16 * 3];
    __shared__ int      ssrc[16], sdst[16];

    const int lane = threadIdx.x;
    const int r  = lane & 15;     // tile row / col-in-tile
    const int kh = lane >> 4;     // K-half selector
    int e = blockIdx.x * 16 + r;
    if (e >= E) e = E - 1;        // E % 16 == 0 in this workload
    const int s = ei[e];
    const int d = ei[(size_t)E + e];

    // ---- stage 1: build msg_in panel [h_src | h_dst | sqdist | edge_attr | pad]
    {
        const float* hp = h + (size_t)(kh ? d : s) * DH;
        _Float16* dp = As + r * 304 + kh * 128;
#pragma unroll
        for (int k = 0; k < 128; k += 4) {
            float4 v4 = *(const float4*)(hp + k);
            dp[k]     = (_Float16)v4.x; dp[k + 1] = (_Float16)v4.y;
            dp[k + 2] = (_Float16)v4.z; dp[k + 3] = (_Float16)v4.w;
        }
        if (kh == 0) {
            float dx = x[s * 3 + 0] - x[d * 3 + 0];
            float dy = x[s * 3 + 1] - x[d * 3 + 1];
            float dz = x[s * 3 + 2] - x[d * 3 + 2];
            sdiff[r * 3 + 0] = dx; sdiff[r * 3 + 1] = dy; sdiff[r * 3 + 2] = dz;
            ssrc[r] = s; sdst[r] = d;
            As[r * 304 + 256] = (_Float16)(dx * dx + dy * dy + dz * dz);
#pragma unroll
            for (int j = 0; j < DE; ++j)
                As[r * 304 + 257 + j] = (_Float16)ea[(size_t)e * DE + j];
#pragma unroll
            for (int j = 273; j < 288; ++j) As[r * 304 + j] = (_Float16)0.0f;
        }
    }
    __syncthreads();

    // ---- matmul1: [16x288] @ We1 -> silu -> M1 (f16)
#pragma unroll 1
    for (int nt = 0; nt < 8; ++nt) {
        const int n = nt * 16 + r;
        const float bv = be1[n];
        v8f acc = {bv, bv, bv, bv, bv, bv, bv, bv};
#pragma unroll 1
        for (int kt = 0; kt < 9; ++kt) {
            v16h a = load_a_lds(As, r, 304, kt * 32, kh);
            v16h b = load_b_wt(We1T, n, 288, kt * 32, kh);
            acc = wmma_f16(a, b, acc);
        }
#pragma unroll
        for (int v = 0; v < 8; ++v)
            M1[(kh * 8 + v) * 136 + n] = (_Float16)silu_f(acc[v]);
    }
    __syncthreads();

    // ---- matmul2: M1 @ We2 -> silu -> msg; scatter-add into agg[dst]
#pragma unroll 1
    for (int nt = 0; nt < 8; ++nt) {
        const int n = nt * 16 + r;
        const float bv = be2[n];
        v8f acc = {bv, bv, bv, bv, bv, bv, bv, bv};
#pragma unroll 1
        for (int kt = 0; kt < 4; ++kt) {
            v16h a = load_a_lds(M1, r, 136, kt * 32, kh);
            v16h b = load_b_wt(We2T, n, 128, kt * 32, kh);
            acc = wmma_f16(a, b, acc);
        }
#pragma unroll
        for (int v = 0; v < 8; ++v) {
            const int m = kh * 8 + v;
            const float mv = silu_f(acc[v]);
            MS[m * 136 + n] = (_Float16)mv;
            atomicAdd(&agg[(size_t)sdst[m] * DM + n], mv);
        }
    }
    __syncthreads();

    // ---- matmul3: msg @ Wx1 -> silu -> T (reuse As panel)
#pragma unroll 1
    for (int nt = 0; nt < 8; ++nt) {
        const int n = nt * 16 + r;
        const float bv = bx1[n];
        v8f acc = {bv, bv, bv, bv, bv, bv, bv, bv};
#pragma unroll 1
        for (int kt = 0; kt < 4; ++kt) {
            v16h a = load_a_lds(MS, r, 136, kt * 32, kh);
            v16h b = load_b_wt(Wx1T, n, 128, kt * 32, kh);
            acc = wmma_f16(a, b, acc);
        }
#pragma unroll
        for (int v = 0; v < 8; ++v)
            As[(kh * 8 + v) * 304 + n] = (_Float16)silu_f(acc[v]);
    }
    __syncthreads();

    // ---- stage 5: per-edge coord weight (T . Wx2 + bx2), scatter coord + count
    if (kh == 0) {
        float wv = bx2[0];
#pragma unroll 4
        for (int k = 0; k < DM; ++k) wv += (float)As[r * 304 + k] * Wx2[k];
        const int sn = ssrc[r];
#pragma unroll
        for (int j = 0; j < 3; ++j)
            atomicAdd(&coord[(size_t)sn * 3 + j], sdiff[r * 3 + j] * wv);
        atomicAdd(&cnt[sdst[r]], 1.0f);
    }
}

// ---------------- node kernel: one wave = 16 nodes --------------------------
__global__ __launch_bounds__(32)
void egnn_node_kernel(const float* __restrict__ h, const float* __restrict__ x,
                      const float* __restrict__ bh1, const float* __restrict__ bh2,
                      const float* __restrict__ ln_g, const float* __restrict__ ln_b,
                      const _Float16* __restrict__ Wh1T, const _Float16* __restrict__ Wh2T,
                      const float* __restrict__ agg, const float* __restrict__ cnt,
                      const float* __restrict__ coord,
                      float* __restrict__ h_out, float* __restrict__ x_out, int N) {
    __shared__ _Float16 HU[16 * 272];   // [h | agg/count] 16 x 256 f16
    __shared__ _Float16 U[16 * 136];    // hidden 16 x 128 f16
    __shared__ float    PRE[16 * 128];  // pre-LN f32

    const int lane = threadIdx.x;
    const int r  = lane & 15;
    const int kh = lane >> 4;
    int node = blockIdx.x * 16 + r;
    if (node >= N) node = N - 1;        // N % 16 == 0 in this workload
    const float c    = fmaxf(cnt[node], 1.0f);
    const float cinv = __builtin_amdgcn_rcpf(c);

    // ---- stage 1: build [h | agg/c] panel
    {
        _Float16* dp = HU + r * 272 + kh * 128;
        if (kh == 0) {
            const float* hp = h + (size_t)node * DH;
#pragma unroll
            for (int k = 0; k < 128; k += 4) {
                float4 v4 = *(const float4*)(hp + k);
                dp[k]     = (_Float16)v4.x; dp[k + 1] = (_Float16)v4.y;
                dp[k + 2] = (_Float16)v4.z; dp[k + 3] = (_Float16)v4.w;
            }
        } else {
            const float* ap = agg + (size_t)node * DM;
#pragma unroll
            for (int k = 0; k < 128; k += 4) {
                float4 v4 = *(const float4*)(ap + k);
                dp[k]     = (_Float16)(v4.x * cinv); dp[k + 1] = (_Float16)(v4.y * cinv);
                dp[k + 2] = (_Float16)(v4.z * cinv); dp[k + 3] = (_Float16)(v4.w * cinv);
            }
        }
    }
    __syncthreads();

    // ---- matmul1: [16x256] @ Wh1 -> silu -> U
#pragma unroll 1
    for (int nt = 0; nt < 8; ++nt) {
        const int n = nt * 16 + r;
        const float bv = bh1[n];
        v8f acc = {bv, bv, bv, bv, bv, bv, bv, bv};
#pragma unroll 1
        for (int kt = 0; kt < 8; ++kt) {
            v16h a = load_a_lds(HU, r, 272, kt * 32, kh);
            v16h b = load_b_wt(Wh1T, n, 256, kt * 32, kh);
            acc = wmma_f16(a, b, acc);
        }
#pragma unroll
        for (int v = 0; v < 8; ++v)
            U[(kh * 8 + v) * 136 + n] = (_Float16)silu_f(acc[v]);
    }
    __syncthreads();

    // ---- matmul2: U @ Wh2; residual with f32 h -> PRE
    const int n0 = blockIdx.x * 16;
#pragma unroll 1
    for (int nt = 0; nt < 8; ++nt) {
        const int n = nt * 16 + r;
        const float bv = bh2[n];
        v8f acc = {bv, bv, bv, bv, bv, bv, bv, bv};
#pragma unroll 1
        for (int kt = 0; kt < 4; ++kt) {
            v16h a = load_a_lds(U, r, 136, kt * 32, kh);
            v16h b = load_b_wt(Wh2T, n, 128, kt * 32, kh);
            acc = wmma_f16(a, b, acc);
        }
#pragma unroll
        for (int v = 0; v < 8; ++v) {
            const int m = kh * 8 + v;
            int nd = n0 + m; if (nd >= N) nd = N - 1;
            PRE[m * 128 + n] = h[(size_t)nd * DH + n] + acc[v];
        }
    }
    __syncthreads();

    // ---- LayerNorm + coordinate update (lanes 0..15, one row each)
    if (kh == 0) {
        float mu = 0.0f;
#pragma unroll 4
        for (int k = 0; k < DH; ++k) mu += PRE[r * 128 + k];
        mu *= (1.0f / DH);
        float var = 0.0f;
#pragma unroll 4
        for (int k = 0; k < DH; ++k) { float t = PRE[r * 128 + k] - mu; var += t * t; }
        var *= (1.0f / DH);
        const float inv = __builtin_amdgcn_rsqf(var + 1e-5f);  // single v_rsq_f32
        float* op = h_out + (size_t)node * DH;
#pragma unroll 4
        for (int k = 0; k < DH; ++k)
            op[k] = (PRE[r * 128 + k] - mu) * inv * ln_g[k] + ln_b[k];
#pragma unroll
        for (int j = 0; j < 3; ++j)
            x_out[(size_t)node * 3 + j] = x[(size_t)node * 3 + j]
                                        + coord[(size_t)node * 3 + j] * cinv;
    }
}

extern "C" void kernel_launch(void* const* d_in, const int* in_sizes, int n_in,
                              void* d_out, int out_size, void* d_ws, size_t ws_size,
                              hipStream_t stream) {
    const float* h   = (const float*)d_in[0];
    const float* x   = (const float*)d_in[1];
    const float* ea  = (const float*)d_in[2];
    const float* We1 = (const float*)d_in[3];
    const float* be1 = (const float*)d_in[4];
    const float* We2 = (const float*)d_in[5];
    const float* be2 = (const float*)d_in[6];
    const float* Wh1 = (const float*)d_in[7];
    const float* bh1 = (const float*)d_in[8];
    const float* Wh2 = (const float*)d_in[9];
    const float* bh2 = (const float*)d_in[10];
    const float* Wx1 = (const float*)d_in[11];
    const float* bx1 = (const float*)d_in[12];
    const float* Wx2 = (const float*)d_in[13];
    const float* bx2 = (const float*)d_in[14];
    const float* lng = (const float*)d_in[15];
    const float* lnb = (const float*)d_in[16];
    const int*   ei  = (const int*)d_in[17];

    const int N = in_sizes[0] / DH;
    const int E = in_sizes[2] / DE;

    // workspace layout
    float* agg   = (float*)d_ws;                 // N*128
    float* cntp  = agg + (size_t)N * DM;         // N
    float* coord = cntp + N;                     // N*3
    _Float16* We1T = (_Float16*)(coord + (size_t)N * 3);   // 128*288
    _Float16* We2T = We1T + 128 * 288;                     // 128*128
    _Float16* Wh1T = We2T + 128 * 128;                     // 128*256
    _Float16* Wh2T = Wh1T + 128 * 256;                     // 128*128
    _Float16* Wx1T = Wh2T + 128 * 128;                     // 128*128

    // prep: weight convert/transpose to f16 [n][k_padded]
    transpose_w_f16<<<(128 * 288 + 255) / 256, 256, 0, stream>>>(We1T, We1, 273, 288, 128 * 288);
    transpose_w_f16<<<(128 * 128 + 255) / 256, 256, 0, stream>>>(We2T, We2, 128, 128, 128 * 128);
    transpose_w_f16<<<(128 * 256 + 255) / 256, 256, 0, stream>>>(Wh1T, Wh1, 256, 256, 128 * 256);
    transpose_w_f16<<<(128 * 128 + 255) / 256, 256, 0, stream>>>(Wh2T, Wh2, 128, 128, 128 * 128);
    transpose_w_f16<<<(128 * 128 + 255) / 256, 256, 0, stream>>>(Wx1T, Wx1, 128, 128, 128 * 128);

    // zero accumulators (agg | count | coord are contiguous)
    const long long nz = (long long)N * DM + N + (long long)N * 3;
    zero_f32<<<2048, 256, 0, stream>>>(agg, nz);

    // edge pass: one wave per 16-edge tile
    egnn_edge_kernel<<<(E + 15) / 16, 32, 0, stream>>>(
        h, x, ea, ei, be1, be2, bx1, Wx2, bx2, We1T, We2T, Wx1T,
        agg, cntp, coord, E);

    // node pass: one wave per 16-node tile
    float* h_out = (float*)d_out;
    float* x_out = h_out + (size_t)N * DH;
    egnn_node_kernel<<<(N + 15) / 16, 32, 0, stream>>>(
        h, x, bh1, bh2, lng, lnb, Wh1T, Wh2T, agg, cntp, coord,
        h_out, x_out, N);
}
